// SelfAttention_70257075028552
// MI455X (gfx1250) — matrix operand order repaired and checked by reading, not compiled, not run on hardware
//
#include <hip/hip_runtime.h>

// ---------------------------------------------------------------------------
// Self-attention for MI455X (gfx1250): bf16 WMMA GEMMs (async global->LDS
// double-buffered staging) + fused flash attention
// ---------------------------------------------------------------------------

typedef __bf16 bf16_t;
typedef __attribute__((ext_vector_type(16))) __bf16        v16bf;
typedef __attribute__((ext_vector_type(8)))  float         v8f;
typedef __attribute__((ext_vector_type(4)))  unsigned int  u32x4;

union FragBF {
    v16bf v;
    u32x4 u[2];
};

constexpr int Bsz = 2;
constexpr int Tsz = 2048;
constexpr int Ksz = 1024;
constexpr int Hsz = 16;
constexpr int Dh  = 64;
constexpr int Mrows = Bsz * Tsz;   // 4096

// ---------------------------------------------------------------------------
// f32 -> bf16 conversion (grid-stride)
// ---------------------------------------------------------------------------
__global__ void f32_to_bf16_kernel(const float* __restrict__ in,
                                   bf16_t* __restrict__ out, long n) {
    long i = (long)blockIdx.x * blockDim.x + threadIdx.x;
    long stride = (long)gridDim.x * blockDim.x;
    for (; i < n; i += stride) out[i] = (bf16_t)in[i];
}

// ---------------------------------------------------------------------------
// GEMM: out[m,n] = sum_k A[m,k] * W[n,k]   (i.e. A @ W^T)
// A: [M,Kd] bf16 row-major.  W: [N,Kd] bf16 row-major.
// Block = 128 threads (4 waves, 2x2), block tile 128x128, wave tile 64x64.
// K-slices (128x32 of A and of W) staged into LDS with double-buffered
// GLOBAL_LOAD_ASYNC_TO_LDS_B128 (ASYNCcnt), fragments read with ds_load_b128.
// MODE 0: bf16 out, natural [M,N]
// MODE 1: bf16 out, V-transposed [b,h,d,t] (packed b128 stores)
// MODE 2: f32 out + bias
// ---------------------------------------------------------------------------
template <int MODE>
__global__ __launch_bounds__(128)
void gemm_bf16_wmma(const bf16_t* __restrict__ A, const bf16_t* __restrict__ W,
                    bf16_t* __restrict__ outb, float* __restrict__ outf,
                    const float* __restrict__ bias, int M, int N, int Kd) {
    __shared__ __align__(16) bf16_t As[2][128][32];
    __shared__ __align__(16) bf16_t Ws[2][128][32];

    const int tid  = threadIdx.x;
    const int lane = tid & 31;
    const int wid  = tid >> 5;
    const int wm   = wid >> 1;
    const int wn   = wid & 1;
    const int mblk = blockIdx.y * 128;
    const int nblk = blockIdx.x * 128;
    const int m0   = mblk + wm * 64;
    const int n0   = nblk + wn * 64;
    const int lh   = lane & 15;        // position within 16-lane half
    const int hi   = lane >> 4;        // which half

    // Async-stage one 128x32 k-slice of A and of W into LDS buffer `buf`.
    // Thread `tid` owns global row (mblk+tid)/(nblk+tid): 64 bytes = 4 b128 chunks.
    auto stage = [&](int buf, int kb) {
        const bf16_t* ga = A + (size_t)(mblk + tid) * Kd + kb;
        const bf16_t* gw = W + (size_t)(nblk + tid) * Kd + kb;
        unsigned la = (unsigned)(uintptr_t)(&As[buf][tid][0]);
        unsigned lw = (unsigned)(uintptr_t)(&Ws[buf][tid][0]);
        #pragma unroll
        for (int c = 0; c < 4; ++c) {
            asm volatile("global_load_async_to_lds_b128 %0, %1, off"
                         :: "v"(la + 16 * c), "v"(ga + 8 * c) : "memory");
            asm volatile("global_load_async_to_lds_b128 %0, %1, off"
                         :: "v"(lw + 16 * c), "v"(gw + 8 * c) : "memory");
        }
    };

    v8f acc[4][4];
    for (int i = 0; i < 4; ++i)
        for (int j = 0; j < 4; ++j)
            for (int r = 0; r < 8; ++r) acc[i][j][r] = 0.0f;

    stage(0, 0);
    asm volatile("s_wait_asynccnt 0x0" ::: "memory");
    __syncthreads();

    for (int kb = 0; kb < Kd; kb += 32) {
        const int buf = (kb >> 5) & 1;
        if (kb + 32 < Kd) stage(buf ^ 1, kb + 32);   // overlap DMA with compute

        FragBF fa[4], fb[4];
        // A fragments: 16x32 bf16. lane<16: rows m, K 0..15; lane>=16: K 16..31
        #pragma unroll
        for (int i = 0; i < 4; ++i) {
            const bf16_t* p = &As[buf][wm * 64 + 16 * i + lh][hi << 4];
            fa[i].u[0] = *(const u32x4*)p;
            fa[i].u[1] = *(const u32x4*)(p + 8);
        }
        // B fragments: 32x16 bf16, B[kk,n] = W[n,kk]. lane<16: K 0..15; lane>=16: K 16..31
        #pragma unroll
        for (int j = 0; j < 4; ++j) {
            const bf16_t* p = &Ws[buf][wn * 64 + 16 * j + lh][hi << 4];
            fb[j].u[0] = *(const u32x4*)p;
            fb[j].u[1] = *(const u32x4*)(p + 8);
        }
        #pragma unroll
        for (int i = 0; i < 4; ++i)
            #pragma unroll
            for (int j = 0; j < 4; ++j)
                acc[i][j] = __builtin_amdgcn_wmma_f32_16x16x32_bf16(
                    false, fa[i].v, false, fb[j].v, (short)0, acc[i][j], false, false);

        if (kb + 32 < Kd)
            asm volatile("s_wait_asynccnt 0x0" ::: "memory");
        __syncthreads();
    }

    // Epilogue. C layout: VGPR r -> row r (lanes 0-15) or 8+r (lanes 16-31); col = lh.
    #pragma unroll
    for (int i = 0; i < 4; ++i) {
        #pragma unroll
        for (int j = 0; j < 4; ++j) {
            const int col = n0 + 16 * j + lh;
            if (MODE == 1) {
                // 8 consecutive t values per lane -> one packed 16B store
                const int h = col >> 6, d = col & 63;
                const int b = (m0 + 16 * i) >> 11;
                const int t0 = (m0 + 16 * i + (hi << 3)) & 2047;
                union { bf16_t e[8]; u32x4 q; } pk;
                #pragma unroll
                for (int r = 0; r < 8; ++r) pk.e[r] = (bf16_t)acc[i][j][r];
                *(u32x4*)(outb + ((size_t)(b * Hsz + h) * Dh + d) * Tsz + t0) = pk.q;
            } else {
                #pragma unroll
                for (int r = 0; r < 8; ++r) {
                    const int row = m0 + 16 * i + r + (hi << 3);
                    float val = acc[i][j][r];
                    if (MODE == 0) {
                        outb[(size_t)row * N + col] = (bf16_t)val;
                    } else {
                        outf[(size_t)row * N + col] = val + bias[col];
                    }
                }
            }
        }
    }
}

// ---------------------------------------------------------------------------
// Fused flash attention.
// Q, Km: [B*T, 1024] bf16 (head h at column offset h*64).
// Vt:    [B, H, 64, T] bf16 (V transposed so PV B-fragments are contiguous).
// mask:  [B, T] int32 (key mask).
// ys:    [B*T, 1024] bf16.
// Block = 128 threads, 4 waves; each wave owns 16 query rows; 64-key blocks.
// ---------------------------------------------------------------------------
__global__ __launch_bounds__(128)
void attn_wmma(const bf16_t* __restrict__ Q, const bf16_t* __restrict__ Km,
               const bf16_t* __restrict__ Vt, const int* __restrict__ mask,
               bf16_t* __restrict__ ys) {
    __shared__ __align__(16) bf16_t Pt[4][16][64];   // per-wave P tile bounce

    const int lane = threadIdx.x & 31;
    const int wid  = threadIdx.x >> 5;
    const int lh   = lane & 15;
    const int hi   = lane >> 4;
    const int h    = blockIdx.y;
    const int b    = blockIdx.z;
    const int q0   = blockIdx.x * 64 + wid * 16;
    const int hofs = h * Dh;

    // Q fragments (A-matrix), one per 32-wide slice of head dim
    FragBF qa[2];
    #pragma unroll
    for (int s = 0; s < 2; ++s) {
        const bf16_t* p = Q + (size_t)(b * Tsz + q0 + lh) * Ksz + hofs + 32 * s + (hi << 4);
        qa[s].u[0] = *(const u32x4*)p;
        qa[s].u[1] = *(const u32x4*)(p + 8);
    }

    float mst[8], lst[8];
    v8f oacc[4];
    for (int r = 0; r < 8; ++r) { mst[r] = -1e30f; lst[r] = 0.0f; }
    for (int nb = 0; nb < 4; ++nb)
        for (int r = 0; r < 8; ++r) oacc[nb][r] = 0.0f;

    for (int kb0 = 0; kb0 < Tsz; kb0 += 64) {
        // ---- S = Q @ K^T for this 16x64 tile ----
        v8f sacc[4];
        for (int j = 0; j < 4; ++j)
            for (int r = 0; r < 8; ++r) sacc[j][r] = 0.0f;
        #pragma unroll
        for (int s = 0; s < 2; ++s) {
            #pragma unroll
            for (int j = 0; j < 4; ++j) {
                const bf16_t* p = Km + (size_t)(b * Tsz + kb0 + 16 * j + lh) * Ksz
                                     + hofs + 32 * s + (hi << 4);
                FragBF kf;
                kf.u[0] = *(const u32x4*)p;
                kf.u[1] = *(const u32x4*)(p + 8);
                sacc[j] = __builtin_amdgcn_wmma_f32_16x16x32_bf16(
                    false, qa[s].v, false, kf.v, (short)0, sacc[j], false, false);
            }
        }
        // ---- key mask ----
        #pragma unroll
        for (int j = 0; j < 4; ++j) {
            const int mv = mask[b * Tsz + kb0 + 16 * j + lh];
            if (mv == 0) {
                #pragma unroll
                for (int r = 0; r < 8; ++r) sacc[j][r] = -1e30f;
            }
        }
        // ---- online softmax: row max (reduce across the 16-lane half) ----
        float rmax[8];
        #pragma unroll
        for (int r = 0; r < 8; ++r) {
            float m = fmaxf(fmaxf(sacc[0][r], sacc[1][r]), fmaxf(sacc[2][r], sacc[3][r]));
            #pragma unroll
            for (int off = 1; off < 16; off <<= 1)
                m = fmaxf(m, __shfl_xor(m, off, 32));
            rmax[r] = m;
        }
        float mnew[8], alpha[8], rsum[8];
        #pragma unroll
        for (int r = 0; r < 8; ++r) {
            mnew[r]  = fmaxf(mst[r], rmax[r]);
            alpha[r] = __expf(mst[r] - mnew[r]);
            mst[r]   = mnew[r];
            rsum[r]  = 0.0f;
        }
        // ---- P = exp(S - m), store bf16 to LDS in plain row-major ----
        #pragma unroll
        for (int j = 0; j < 4; ++j) {
            #pragma unroll
            for (int r = 0; r < 8; ++r) {
                const float sv  = sacc[j][r];
                const float pij = (sv < -5e29f) ? 0.0f : __expf(sv - mnew[r]);
                rsum[r] += pij;
                Pt[wid][r + (hi << 3)][16 * j + lh] = (bf16_t)pij;
            }
        }
        #pragma unroll
        for (int r = 0; r < 8; ++r) {
            #pragma unroll
            for (int off = 1; off < 16; off <<= 1)
                rsum[r] += __shfl_xor(rsum[r], off, 32);
            lst[r] = lst[r] * alpha[r] + rsum[r];
        }
        // rescale running output
        #pragma unroll
        for (int nb = 0; nb < 4; ++nb)
            #pragma unroll
            for (int r = 0; r < 8; ++r) oacc[nb][r] *= alpha[r];

        __syncthreads();

        // ---- O += P @ V  (P A-fragments from LDS; V B-fragments contiguous in Vt) ----
        #pragma unroll
        for (int s = 0; s < 2; ++s) {
            FragBF pa;
            const bf16_t* pp = &Pt[wid][lh][32 * s + (hi << 4)];
            pa.u[0] = *(const u32x4*)pp;
            pa.u[1] = *(const u32x4*)(pp + 8);
            #pragma unroll
            for (int nb = 0; nb < 4; ++nb) {
                const bf16_t* pv = Vt + ((size_t)(b * Hsz + h) * Dh + 16 * nb + lh) * Tsz
                                      + kb0 + 32 * s + (hi << 4);
                FragBF vf;
                vf.u[0] = *(const u32x4*)pv;
                vf.u[1] = *(const u32x4*)(pv + 8);
                oacc[nb] = __builtin_amdgcn_wmma_f32_16x16x32_bf16(
                    false, pa.v, false, vf.v, (short)0, oacc[nb], false, false);
            }
        }
        __syncthreads();
    }

    // ---- normalize and write ys ----
    float inv[8];
    #pragma unroll
    for (int r = 0; r < 8; ++r) inv[r] = 1.0f / (lst[r] + 1e-20f);
    #pragma unroll
    for (int nb = 0; nb < 4; ++nb) {
        const int d = 16 * nb + lh;
        #pragma unroll
        for (int r = 0; r < 8; ++r) {
            const int q = q0 + r + (hi << 3);
            ys[(size_t)(b * Tsz + q) * Ksz + hofs + d] = (bf16_t)(oacc[nb][r] * inv[r]);
        }
    }
}

// ---------------------------------------------------------------------------
// Host launcher
// ---------------------------------------------------------------------------
extern "C" void kernel_launch(void* const* d_in, const int* in_sizes, int n_in,
                              void* d_out, int out_size, void* d_ws, size_t ws_size,
                              hipStream_t stream) {
    const float* x    = (const float*)d_in[0];
    const int*   mask = (const int*)d_in[1];
    const float* Wk   = (const float*)d_in[2];
    const float* Wq   = (const float*)d_in[3];
    const float* Wv   = (const float*)d_in[4];
    const float* Wu   = (const float*)d_in[5];
    const float* bu   = (const float*)d_in[6];
    float* out = (float*)d_out;

    constexpr size_t XN = (size_t)Mrows * Ksz;   // 4 M elements
    constexpr size_t WN = (size_t)Ksz * Ksz;     // 1 M elements

    char* ws = (char*)d_ws;
    bf16_t* xb   = (bf16_t*)ws;            ws += XN * 2;
    bf16_t* wkb  = (bf16_t*)ws;            ws += WN * 2;
    bf16_t* wqb  = (bf16_t*)ws;            ws += WN * 2;
    bf16_t* wvb  = (bf16_t*)ws;            ws += WN * 2;
    bf16_t* wub  = (bf16_t*)ws;            ws += WN * 2;
    bf16_t* kbuf = (bf16_t*)ws;            ws += XN * 2;
    bf16_t* qbuf = (bf16_t*)ws;            ws += XN * 2;
    bf16_t* vtb  = (bf16_t*)ws;            ws += XN * 2;   // [B,H,64,T]
    bf16_t* ysb  = (bf16_t*)ws;            ws += XN * 2;

    // f32 -> bf16 conversions
    {
        int blk = 256;
        f32_to_bf16_kernel<<<(int)((XN + blk - 1) / blk), blk, 0, stream>>>(x,  xb,  (long)XN);
        f32_to_bf16_kernel<<<(int)((WN + blk - 1) / blk), blk, 0, stream>>>(Wk, wkb, (long)WN);
        f32_to_bf16_kernel<<<(int)((WN + blk - 1) / blk), blk, 0, stream>>>(Wq, wqb, (long)WN);
        f32_to_bf16_kernel<<<(int)((WN + blk - 1) / blk), blk, 0, stream>>>(Wv, wvb, (long)WN);
        f32_to_bf16_kernel<<<(int)((WN + blk - 1) / blk), blk, 0, stream>>>(Wu, wub, (long)WN);
    }

    dim3 gg(Ksz / 128, Mrows / 128);   // (8, 32)

    // K, Q projections (natural layout); V projection (transposed store)
    gemm_bf16_wmma<0><<<gg, 128, 0, stream>>>(xb, wkb, kbuf, nullptr, nullptr, Mrows, Ksz, Ksz);
    gemm_bf16_wmma<0><<<gg, 128, 0, stream>>>(xb, wqb, qbuf, nullptr, nullptr, Mrows, Ksz, Ksz);
    gemm_bf16_wmma<1><<<gg, 128, 0, stream>>>(xb, wvb, vtb,  nullptr, nullptr, Mrows, Ksz, Ksz);

    // Fused masked-softmax attention
    attn_wmma<<<dim3(Tsz / 64, Hsz, Bsz), 128, 0, stream>>>(qbuf, kbuf, vtb, mask, ysb);

    // Output projection + bias -> f32 d_out
    gemm_bf16_wmma<2><<<gg, 128, 0, stream>>>(ysb, wub, nullptr, out, bu, Mrows, Ksz, Ksz);
}
